// ExpertsAOQuantizable_6605659701457
// MI455X (gfx1250) — matrix operand order repaired and checked by reading, not compiled, not run on hardware
//
#include <hip/hip_runtime.h>
#include <cstdint>

// ---------------- problem constants ----------------
#define NUM_EXPERTS 8
#define HIDDEN      2048
#define EXPERT_D    1408
#define TWO_D       2816
#define NUM_TOKENS  8192
#define TOPK        2
#define SLOTS       (NUM_TOKENS * TOPK)          // 16384
#define MTILE       32
#define CAP         (SLOTS + NUM_EXPERTS * MTILE)  // padded capacity 16640
#define NTHREADS    512
#define NWAVES      16

// ws int layout: [0..7] counts | [8..16] aligned offsets (offs[8]=total) |
//                [17..24] cursors | [32..32+CAP) perm | then CAP floats of weights
#define WS_PERM  32
#define WS_WTS   (WS_PERM + CAP)
#define WS_BT_OFF (256 * 1024)                   // byte offset of bf16 weights

typedef __attribute__((ext_vector_type(16))) __bf16 v16bf;
typedef __attribute__((ext_vector_type(8)))  __bf16 v8bf;
typedef __attribute__((ext_vector_type(8)))  float  v8f;

// ---------------- routing kernels ----------------
__global__ void k_init(float* __restrict__ out, int out_n, int* __restrict__ ints) {
    int i = blockIdx.x * blockDim.x + threadIdx.x;
    int stride = gridDim.x * blockDim.x;
    for (int j = i; j < out_n; j += stride) out[j] = 0.0f;
    if (i < 32) ints[i] = 0;
    for (int j = i; j < CAP; j += stride) ints[WS_PERM + j] = -1;
}

__global__ void k_hist(const int* __restrict__ idx, int* __restrict__ ints) {
    int i = blockIdx.x * blockDim.x + threadIdx.x;
    if (i < SLOTS) {
        int e = idx[i] & (NUM_EXPERTS - 1);
        atomicAdd(&ints[e], 1);
    }
}

__global__ void k_scan(int* __restrict__ ints) {
    if (threadIdx.x == 0 && blockIdx.x == 0) {
        int off = 0;
        for (int e = 0; e < NUM_EXPERTS; ++e) {
            ints[8 + e] = off;
            off += (ints[e] + (MTILE - 1)) & ~(MTILE - 1);  // 32-align each segment
        }
        ints[16] = off;   // padded total
    }
}

__global__ void k_scatter(const int* __restrict__ idx, const float* __restrict__ scores,
                          int* __restrict__ ints, float* __restrict__ wts) {
    int i = blockIdx.x * blockDim.x + threadIdx.x;
    if (i < SLOTS) {
        int e = idx[i] & (NUM_EXPERTS - 1);
        int p = atomicAdd(&ints[17 + e], 1);
        int slot = ints[8 + e] + p;
        ints[WS_PERM + slot] = i >> 1;      // token id
        wts[slot] = scores[i];
    }
}

// ---------------- weight transpose + bf16 quantize ----------------
// in: [E][R][C] fp32 row-major  ->  out: [E][C][R] bf16 row-major
__global__ __launch_bounds__(256) void k_transpose(const float* __restrict__ in,
                                                   __bf16* __restrict__ outp,
                                                   int R, int C) {
    __shared__ float tile[32][33];
    const int e  = blockIdx.z;
    const int c0 = blockIdx.x * 32;
    const int r0 = blockIdx.y * 32;
    const float* src = in   + (size_t)e * R * C;
    __bf16*      dst = outp + (size_t)e * R * C;
    const int tx = threadIdx.x & 31;
    const int ty = threadIdx.x >> 5;
#pragma unroll
    for (int i = ty; i < 32; i += 8)
        tile[i][tx] = src[(size_t)(r0 + i) * C + c0 + tx];
    __syncthreads();
#pragma unroll
    for (int i = ty; i < 32; i += 8)
        dst[(size_t)(c0 + i) * R + r0 + tx] = (__bf16)tile[tx][i];
}

// ---------------- WMMA fragment helpers ----------------
// A (16-bit, 16x32): lanes 0-15 -> M=lane, lanes 16-31 -> M=lane-16, K base +16.
// elems: VGPR0..3 = K[base..base+7], VGPR4..7 = K[base+8..base+15].
__device__ __forceinline__ v16bf load_a_frag(const __bf16* __restrict__ lds,
                                             int ldk, int m, int kbase) {
    const __bf16* p = lds + m * ldk + kbase;
    v8bf lo = *(const v8bf*)(p);
    v8bf hi = *(const v8bf*)(p + 8);
    return __builtin_shufflevector(lo, hi, 0,1,2,3,4,5,6,7,8,9,10,11,12,13,14,15);
}

// B (16-bit, 32x16): lane -> N = lane&15; lanes 0-15 hold K[k0..k0+15],
// lanes 16-31 hold K[k0+16..k0+31].

// Fast path: B^T stored bf16 row-major [N][ldk] -> 32 contiguous bytes per lane.
__device__ __forceinline__ v16bf load_b_frag_bt(const __bf16* __restrict__ Bt,
                                                int ldk, int kbase, int n) {
    return *(const v16bf*)(Bt + (size_t)n * ldk + kbase);
}

// Fallback: B fp32 row-major [K][ldn], convert on the fly.
__device__ __forceinline__ v16bf load_b_frag_f32(const float* __restrict__ B,
                                                 int ldn, int kbase, int n) {
    v16bf b;
    const float* p = B + (size_t)kbase * ldn + n;
#pragma unroll
    for (int j = 0; j < 16; ++j) {
        b[j] = (__bf16)(*p);
        p += ldn;
    }
    return b;
}

__device__ __forceinline__ v8f wmma_bf16(v16bf a, v16bf b, v8f c) {
    return __builtin_amdgcn_wmma_f32_16x16x32_bf16(false, a, false, b, (short)0, c,
                                                   false, false);
}

__device__ __forceinline__ float silu_mul(float g, float u) {
    // g * sigmoid(g) * u with fast v_rcp_f32 (avoids IEEE divide chain)
    return g * __builtin_amdgcn_rcpf(1.0f + __expf(-g)) * u;
}

// ---------------- fused expert kernel ----------------
// 512 threads = 16 wave32s; 32-token tile; two M-subtiles share every B fragment
// so the steady state is ~1 global_load_b128 + 1 ds_load_b128 per v_wmma.
template <bool BT>
__global__ __launch_bounds__(NTHREADS) void k_moe(const float* __restrict__ x,
                                                  const float* __restrict__ up_w,
                                                  const float* __restrict__ dn_w,
                                                  const __bf16* __restrict__ upT,
                                                  const __bf16* __restrict__ dnT,
                                                  const int* __restrict__ ints,
                                                  const float* __restrict__ wts,
                                                  float* __restrict__ out) {
    __shared__ alignas(16) __bf16 lds_x[MTILE * HIDDEN];     // 128 KB
    __shared__ alignas(16) __bf16 lds_act[MTILE * EXPERT_D]; //  88 KB
    __shared__ int   s_tok[MTILE];
    __shared__ float s_w[MTILE];
    __shared__ int   s_e;

    const int tile = blockIdx.x * MTILE;
    if (tile >= ints[16]) return;           // beyond padded total

    const int tid = threadIdx.x;
    if (tid == 0) {
        int e = 0;
        while (e < NUM_EXPERTS - 1 && tile >= ints[8 + e + 1]) ++e;
        s_e = e;
    }
    if (tid < MTILE) {
        int t = ints[WS_PERM + tile + tid];
        s_tok[tid] = t;
        s_w[tid] = (t >= 0) ? wts[tile + tid] : 0.0f;
    }
    __syncthreads();
    const int e = s_e;

    // stage x tile (fp32 -> bf16) into LDS
    for (int i = tid; i < MTILE * HIDDEN; i += NTHREADS) {
        int m = i >> 11, c = i & (HIDDEN - 1);
        int t = s_tok[m];
        float v = (t >= 0) ? x[(size_t)t * HIDDEN + c] : 0.0f;
        lds_x[(m << 11) + c] = (__bf16)v;
    }
    __syncthreads();

    const int lane = tid & 31;
    const int wv   = tid >> 5;        // wave 0..15
    const int mrow = lane & 15;
    const int hi   = lane >> 4;       // lane-half selector

    // ---- Phase A: h = x @ up_proj[e]; act = silu(gate) * up ----
    const float*  upW  = up_w + (size_t)e * HIDDEN * TWO_D;
    const __bf16* upWT = upT  + (size_t)e * TWO_D * HIDDEN;   // [2816][2048]
    for (int g = wv; g < EXPERT_D / 16; g += NWAVES) {        // 88 gate tiles
        v8f accg0 = {}, accg1 = {}, accu0 = {}, accu1 = {};
        const int ng = g * 16 + mrow;                 // gate column
        const int nu = ng + EXPERT_D;                 // paired up column
#pragma unroll 4
        for (int k0 = 0; k0 < HIDDEN; k0 += 32) {
            v16bf a0 = load_a_frag(lds_x, HIDDEN, mrow,      k0 + hi * 16);
            v16bf a1 = load_a_frag(lds_x, HIDDEN, mrow + 16, k0 + hi * 16);
            v16bf bg, bu;
            if (BT) {
                bg = load_b_frag_bt(upWT, HIDDEN, k0 + hi * 16, ng);
                bu = load_b_frag_bt(upWT, HIDDEN, k0 + hi * 16, nu);
            } else {
                bg = load_b_frag_f32(upW, TWO_D, k0 + hi * 16, ng);
                bu = load_b_frag_f32(upW, TWO_D, k0 + hi * 16, nu);
            }
            accg0 = wmma_bf16(a0, bg, accg0);
            accu0 = wmma_bf16(a0, bu, accu0);
            accg1 = wmma_bf16(a1, bg, accg1);
            accu1 = wmma_bf16(a1, bu, accu1);
        }
#pragma unroll
        for (int r = 0; r < 8; ++r) {
            int m = r + hi * 8;
            lds_act[m * EXPERT_D + g * 16 + mrow] = (__bf16)silu_mul(accg0[r], accu0[r]);
            lds_act[(m + 16) * EXPERT_D + g * 16 + mrow] =
                (__bf16)silu_mul(accg1[r], accu1[r]);
        }
    }
    __syncthreads();

    // ---- Phase B: y = act @ down_proj[e]; out[token] += w * y ----
    const float*  dnW  = dn_w + (size_t)e * EXPERT_D * HIDDEN;
    const __bf16* dnWT = dnT  + (size_t)e * HIDDEN * EXPERT_D;  // [2048][1408]
    for (int nt = wv; nt < HIDDEN / 16; nt += NWAVES) {   // 128 N tiles
        v8f acc0 = {}, acc1 = {};
        const int n = nt * 16 + mrow;
#pragma unroll 4
        for (int k0 = 0; k0 < EXPERT_D; k0 += 32) {
            v16bf a0 = load_a_frag(lds_act, EXPERT_D, mrow,      k0 + hi * 16);
            v16bf a1 = load_a_frag(lds_act, EXPERT_D, mrow + 16, k0 + hi * 16);
            v16bf b;
            if (BT) b = load_b_frag_bt(dnWT, EXPERT_D, k0 + hi * 16, n);
            else    b = load_b_frag_f32(dnW, HIDDEN, k0 + hi * 16, n);
            acc0 = wmma_bf16(a0, b, acc0);
            acc1 = wmma_bf16(a1, b, acc1);
        }
#pragma unroll
        for (int r = 0; r < 8; ++r) {
            int m = r + hi * 8;
            int t0 = s_tok[m];
            int t1 = s_tok[m + 16];
            if (t0 >= 0) unsafeAtomicAdd(&out[(size_t)t0 * HIDDEN + n], s_w[m] * acc0[r]);
            if (t1 >= 0) unsafeAtomicAdd(&out[(size_t)t1 * HIDDEN + n],
                                         s_w[m + 16] * acc1[r]);
        }
    }
}

// ---------------- host launcher ----------------
extern "C" void kernel_launch(void* const* d_in, const int* in_sizes, int n_in,
                              void* d_out, int out_size, void* d_ws, size_t ws_size,
                              hipStream_t stream) {
    const float* x      = (const float*)d_in[0];
    const int*   idx    = (const int*)d_in[1];
    const float* scores = (const float*)d_in[2];
    const float* up_w   = (const float*)d_in[3];
    const float* dn_w   = (const float*)d_in[4];
    float* out = (float*)d_out;

    int*   ints = (int*)d_ws;
    float* wts  = (float*)d_ws + WS_WTS;

    const size_t up_bytes = (size_t)NUM_EXPERTS * HIDDEN * TWO_D * sizeof(__bf16);
    const size_t dn_bytes = (size_t)NUM_EXPERTS * EXPERT_D * HIDDEN * sizeof(__bf16);
    const bool use_bt = ws_size >= WS_BT_OFF + up_bytes + dn_bytes;
    __bf16* upT = (__bf16*)((char*)d_ws + WS_BT_OFF);
    __bf16* dnT = (__bf16*)((char*)d_ws + WS_BT_OFF + up_bytes);

    k_init<<<1024, 256, 0, stream>>>(out, NUM_TOKENS * HIDDEN, ints);
    k_hist<<<(SLOTS + 255) / 256, 256, 0, stream>>>(idx, ints);
    k_scan<<<1, 32, 0, stream>>>(ints);
    k_scatter<<<(SLOTS + 255) / 256, 256, 0, stream>>>(idx, scores, ints, wts);

    if (use_bt) {
        // up:   [E][2048][2816] f32 -> [E][2816][2048] bf16
        k_transpose<<<dim3(TWO_D / 32, HIDDEN / 32, NUM_EXPERTS), 256, 0, stream>>>(
            up_w, upT, HIDDEN, TWO_D);
        // down: [E][1408][2048] f32 -> [E][2048][1408] bf16
        k_transpose<<<dim3(HIDDEN / 32, EXPERT_D / 32, NUM_EXPERTS), 256, 0, stream>>>(
            dn_w, dnT, EXPERT_D, HIDDEN);
        k_moe<true><<<CAP / MTILE, NTHREADS, 0, stream>>>(x, up_w, dn_w, upT, dnT,
                                                          ints, wts, out);
    } else {
        k_moe<false><<<CAP / MTILE, NTHREADS, 0, stream>>>(x, up_w, dn_w, upT, dnT,
                                                           ints, wts, out);
    }
}
